// GCN_67886253081375
// MI455X (gfx1250) — matrix operand order repaired and checked by reading, not compiled, not run on hardware
//
#include <hip/hip_runtime.h>
#include <hip/hip_bf16.h>
#include <cstdint>
#include <cstddef>

// ---------------------------------------------------------------------------
// Types for CDNA5 WMMA (wave32): 16x16x32 bf16 -> f32 accumulate
// ---------------------------------------------------------------------------
typedef __attribute__((ext_vector_type(16))) __bf16 v16bf;
typedef __attribute__((ext_vector_type(8)))  __bf16 v8bf;
typedef __attribute__((ext_vector_type(8)))  float  v8f;

// A-matrix fragment (16-bit 16x32 layout, ISA 7.12.2):
//   lanes 0-15 : K = {0..7} U {16..23},  lanes 16-31 : K = {8..15} U {24..31}
// p = this lane's bf16 activation row at the k-chunk start (16B aligned).
static __device__ __forceinline__ v16bf load_a_frag(const __bf16* __restrict__ p,
                                                    int half) {
  v8bf lo = *(const v8bf*)(p + half * 8);        // K = 8*half + 0..7
  v8bf hi = *(const v8bf*)(p + 16 + half * 8);   // K = 16 + 8*half + 0..7
  return __builtin_shufflevector(lo, hi, 0, 1, 2, 3, 4, 5, 6, 7,
                                 8, 9, 10, 11, 12, 13, 14, 15);
}

// B-matrix fragment (32x16): lane holds column n = lane&15,
//   lanes 0-15 : K = {0..7} U {16..23},  lanes 16-31 : K = {8..15} U {24..31} (16B aligned).
static __device__ __forceinline__ v16bf load_b_frag(const __bf16* __restrict__ p,
                                                    int half) {
  v8bf lo = *(const v8bf*)(p + half * 8);
  v8bf hi = *(const v8bf*)(p + 16 + half * 8);
  return __builtin_shufflevector(lo, hi, 0, 1, 2, 3, 4, 5, 6, 7,
                                 8, 9, 10, 11, 12, 13, 14, 15);
}

// ---------------------------------------------------------------------------
// Utility kernels
// ---------------------------------------------------------------------------
__global__ void fill_u32_kernel(uint32_t* __restrict__ p, uint32_t v, size_t n) {
  size_t i = (size_t)blockIdx.x * blockDim.x + threadIdx.x;
  if (i < n) p[i] = v;
}

// f32 -> bf16, 8 elements per thread (n8 = n/8)
__global__ void cvt_bf16_kernel(const float* __restrict__ in,
                                __bf16* __restrict__ out, size_t n8) {
  size_t i = (size_t)blockIdx.x * blockDim.x + threadIdx.x;
  if (i >= n8) return;
  const float4* p = (const float4*)(in + i * 8);
  float4 a = p[0];
  float4 b = p[1];
  v8bf o;
  o[0] = (__bf16)a.x; o[1] = (__bf16)a.y; o[2] = (__bf16)a.z; o[3] = (__bf16)a.w;
  o[4] = (__bf16)b.x; o[5] = (__bf16)b.y; o[6] = (__bf16)b.z; o[7] = (__bf16)b.w;
  ((v8bf*)out)[i] = o;
}

// (agg / max(deg,1)) -> bf16, rows of 128 floats, 8 elems/thread
__global__ void cvt_scale_bf16_kernel(const float* __restrict__ agg,
                                      const float* __restrict__ deg,
                                      __bf16* __restrict__ out, size_t n8) {
  size_t i = (size_t)blockIdx.x * blockDim.x + threadIdx.x;
  if (i >= n8) return;
  int row = (int)(i >> 4);  // 16 chunks of 8 per 128-wide row
  float s = 1.0f / fmaxf(deg[row], 1.0f);
  const float4* p = (const float4*)(agg + i * 8);
  float4 a = p[0];
  float4 b = p[1];
  v8bf o;
  o[0] = (__bf16)(a.x * s); o[1] = (__bf16)(a.y * s);
  o[2] = (__bf16)(a.z * s); o[3] = (__bf16)(a.w * s);
  o[4] = (__bf16)(b.x * s); o[5] = (__bf16)(b.y * s);
  o[6] = (__bf16)(b.z * s); o[7] = (__bf16)(b.w * s);
  ((v8bf*)out)[i] = o;
}

__global__ void degree_kernel(const int* __restrict__ dst,
                              float* __restrict__ deg, int E) {
  int e = blockIdx.x * blockDim.x + threadIdx.x;
  if (e < E) atomicAdd(&deg[dst[e]], 1.0f);
}

// scatter-add neighbor features: agg[dst] += h[src]; one thread per (edge, 4 floats)
__global__ void scatter_add_kernel(const float* __restrict__ h,
                                   const int* __restrict__ src,
                                   const int* __restrict__ dst,
                                   float* __restrict__ agg, int E) {
  int idx = blockIdx.x * blockDim.x + threadIdx.x;
  int e = idx >> 5;
  if (e >= E) return;
  int q = idx & 31;
  int s = src[e];
  int d = dst[e];
  const float4 v = ((const float4*)(h + (size_t)s * 128))[q];
  float* o = agg + (size_t)d * 128 + q * 4;
  atomicAdd(o + 0, v.x);
  atomicAdd(o + 1, v.y);
  atomicAdd(o + 2, v.z);
  atomicAdd(o + 3, v.w);
}

// ---------------------------------------------------------------------------
// Fused SAGE layer GEMM (pure bf16 operands, f32 accumulate):
//   out = act( aggBf @ Wl^T + b + hBf @ Wr^T ),  all dims 128, rows = nrows
// One wave per 16x16 output tile; 8 x v_wmma_f32_16x16x32_bf16 per wave.
// Dual-stores f32 (for scatter/pooling) and bf16 (next GEMM's A operand).
// ---------------------------------------------------------------------------
__global__ void sage_gemm_kernel(const __bf16* __restrict__ aggBf,
                                 const __bf16* __restrict__ hBf,
                                 const __bf16* __restrict__ WlBf,
                                 const __bf16* __restrict__ WrBf,
                                 const float* __restrict__ bias,
                                 float* __restrict__ outF,
                                 __bf16* __restrict__ outBf,
                                 int nrows, int do_relu) {
  const int K = 128;
  const int NT = 8;  // 128 output cols / 16
  int wave = blockIdx.x * (blockDim.x >> 5) + (threadIdx.x >> 5);
  int lane = threadIdx.x & 31;
  int tileN = wave % NT;
  int tileM = wave / NT;
  if (tileM * 16 >= nrows) return;  // uniform per wave; EXEC stays all-ones

  int half = lane >> 4;
  int r = lane & 15;
  int m = tileM * 16 + r;  // activation row this lane loads
  int n = tileN * 16 + r;  // output channel / weight row this lane loads

  const __bf16* aggRow = aggBf + (size_t)m * K;
  const __bf16* hRow   = hBf   + (size_t)m * K;
  const __bf16* wlRow  = WlBf  + (size_t)n * K;
  const __bf16* wrRow  = WrBf  + (size_t)n * K;

  v8f c = {0.f, 0.f, 0.f, 0.f, 0.f, 0.f, 0.f, 0.f};
#pragma unroll
  for (int kc = 0; kc < 4; ++kc) {
    int k0 = kc * 32;
    v16bf a0 = load_a_frag(aggRow + k0, half);
    v16bf b0 = load_b_frag(wlRow + k0, half);
    c = __builtin_amdgcn_wmma_f32_16x16x32_bf16(false, a0, false, b0,
                                                (short)0, c, false, false);
    v16bf a1 = load_a_frag(hRow + k0, half);
    v16bf b1 = load_b_frag(wrRow + k0, half);
    c = __builtin_amdgcn_wmma_f32_16x16x32_bf16(false, a1, false, b1,
                                                (short)0, c, false, false);
  }

  float bv = bias[n];
  int col = tileN * 16 + r;
#pragma unroll
  for (int i = 0; i < 8; ++i) {
    int row = tileM * 16 + i + half * 8;  // C/D layout: VGPR i -> M=i (+8 hi half)
    float v = c[i] + bv;
    if (do_relu) v = fmaxf(v, 0.f);
    size_t o = (size_t)row * K + col;
    outF[o] = v;
    if (outBf) outBf[o] = (__bf16)v;
  }
}

// ---------------------------------------------------------------------------
// Pooling: monotone int key for float atomic max (handles negatives, -inf)
// ---------------------------------------------------------------------------
static __device__ __forceinline__ int enc_key(float f) {
  int b = __float_as_int(f);
  return (b >= 0) ? b : (b ^ 0x7FFFFFFF);
}
static __device__ __forceinline__ float dec_key(int k) {
  return __int_as_float((k >= 0) ? k : (k ^ 0x7FFFFFFF));
}
#define NEG_INF_KEY 0x807FFFFFu  // enc_key(-inf)

__global__ void pool_scatter_kernel(const float* __restrict__ h,
                                    const int* __restrict__ batch,
                                    int* __restrict__ maxk,
                                    float* __restrict__ sums,
                                    float* __restrict__ cnt,
                                    int nnodes) {
  int idx = blockIdx.x * blockDim.x + threadIdx.x;
  if (idx >= nnodes * 128) return;
  int nd = idx >> 7;
  int f  = idx & 127;
  int g  = batch[nd];
  float v = h[idx];
  atomicMax(&maxk[g * 128 + f], enc_key(v));
  atomicAdd(&sums[g * 128 + f], v);
  if (f == 0) atomicAdd(&cnt[g], 1.0f);
}

// builds pooled [G, 256] = concat(max, mean) directly in bf16 for the head GEMM
__global__ void build_pooled_kernel(const int* __restrict__ maxk,
                                    const float* __restrict__ sums,
                                    const float* __restrict__ cnt,
                                    __bf16* __restrict__ pooledBf, int G) {
  int idx = blockIdx.x * blockDim.x + threadIdx.x;
  if (idx >= G * 256) return;
  int g = idx >> 8;
  int f = idx & 255;
  float v;
  if (f < 128) {
    v = dec_key(maxk[g * 128 + f]);
    if (!isfinite(v)) v = 0.f;  // empty segment -> 0 (matches reference)
  } else {
    v = sums[g * 128 + (f - 128)] / fmaxf(cnt[g], 1.0f);
  }
  pooledBf[idx] = (__bf16)v;
}

// ---------------------------------------------------------------------------
// Final head: out[G,256] = pooled[G,256] @ Wlin^T + blin  (K = 256)
// ---------------------------------------------------------------------------
__global__ void final_gemm_kernel(const __bf16* __restrict__ pooledBf,
                                  const __bf16* __restrict__ WlinBf,
                                  const float* __restrict__ blin,
                                  float* __restrict__ out, int G) {
  const int K = 256;
  const int NT = 16;  // 256 output cols / 16
  int wave = blockIdx.x * (blockDim.x >> 5) + (threadIdx.x >> 5);
  int lane = threadIdx.x & 31;
  int tileN = wave % NT;
  int tileM = wave / NT;
  if (tileM * 16 >= G) return;

  int half = lane >> 4;
  int r = lane & 15;
  int m = tileM * 16 + r;
  int n = tileN * 16 + r;
  const __bf16* aRow = pooledBf + (size_t)m * K;
  const __bf16* wRow = WlinBf   + (size_t)n * K;

  v8f c = {0.f, 0.f, 0.f, 0.f, 0.f, 0.f, 0.f, 0.f};
#pragma unroll
  for (int kc = 0; kc < 8; ++kc) {
    int k0 = kc * 32;
    v16bf a = load_a_frag(aRow + k0, half);
    v16bf b = load_b_frag(wRow + k0, half);
    c = __builtin_amdgcn_wmma_f32_16x16x32_bf16(false, a, false, b,
                                                (short)0, c, false, false);
  }

  float bv = blin[n];
#pragma unroll
  for (int i = 0; i < 8; ++i) {
    int row = tileM * 16 + i + half * 8;
    out[(size_t)row * 256 + tileN * 16 + r] = c[i] + bv;
  }
}

// ---------------------------------------------------------------------------
// Host-side launch
// ---------------------------------------------------------------------------
extern "C" void kernel_launch(void* const* d_in, const int* in_sizes, int n_in,
                              void* d_out, int out_size, void* d_ws, size_t ws_size,
                              hipStream_t stream) {
  (void)n_in; (void)ws_size;
  const float* x     = (const float*)d_in[0];
  const int*   ei    = (const int*)d_in[1];   // [2, E] flat: src then dst
  const int*   batch = (const int*)d_in[2];
  const float* W1l = (const float*)d_in[3];
  const float* b1  = (const float*)d_in[4];
  const float* W1r = (const float*)d_in[5];
  const float* W2l = (const float*)d_in[6];
  const float* b2  = (const float*)d_in[7];
  const float* W2r = (const float*)d_in[8];
  const float* W3l = (const float*)d_in[9];
  const float* b3  = (const float*)d_in[10];
  const float* W3r = (const float*)d_in[11];
  const float* Wlin = (const float*)d_in[12];
  const float* blin = (const float*)d_in[13];
  float* outp = (float*)d_out;

  const int N = in_sizes[2];          // 50000 nodes (batch length)
  const int E = in_sizes[1] / 2;      // 600000 edges
  const int G = out_size / 256;       // 512 graphs
  const int* src = ei;
  const int* dst = ei + E;
  const size_t featN = (size_t)N * 128;

  // workspace layout (256B aligned regions)
  char* base = (char*)d_ws;
  size_t off = 0;
  auto alloc = [&](size_t bytes) {
    void* p = base + off;
    off = (off + bytes + 255) & ~(size_t)255;
    return p;
  };
  float*  deg    = (float*)alloc((size_t)N * 4);
  float*  agg    = (float*)alloc(featN * 4);
  float*  F0     = (float*)alloc(featN * 4);          // f32 activations (reused)
  __bf16* aggBf  = (__bf16*)alloc(featN * 2);
  __bf16* xBf    = (__bf16*)alloc(featN * 2);         // also reused as H2bf
  __bf16* h1Bf   = (__bf16*)alloc(featN * 2);
  int*    maxk   = (int*)alloc((size_t)G * 128 * 4);
  float*  sums   = (float*)alloc((size_t)G * 128 * 4);
  float*  cnt    = (float*)alloc((size_t)G * 4);
  __bf16* pooledBf = (__bf16*)alloc((size_t)G * 256 * 2);
  __bf16* W1lBf = (__bf16*)alloc(16384 * 2);
  __bf16* W1rBf = (__bf16*)alloc(16384 * 2);
  __bf16* W2lBf = (__bf16*)alloc(16384 * 2);
  __bf16* W2rBf = (__bf16*)alloc(16384 * 2);
  __bf16* W3lBf = (__bf16*)alloc(16384 * 2);
  __bf16* W3rBf = (__bf16*)alloc(16384 * 2);
  __bf16* WlinBf = (__bf16*)alloc(65536 * 2);
  __bf16* h2Bf = xBf;  // lifetime-disjoint alias

  const int T = 256;
  auto blocks = [](size_t n, int t) { return (unsigned)((n + t - 1) / t); };

  // pre-convert weights and input features to bf16
  cvt_bf16_kernel<<<blocks(16384 / 8, T), T, 0, stream>>>(W1l, W1lBf, 16384 / 8);
  cvt_bf16_kernel<<<blocks(16384 / 8, T), T, 0, stream>>>(W1r, W1rBf, 16384 / 8);
  cvt_bf16_kernel<<<blocks(16384 / 8, T), T, 0, stream>>>(W2l, W2lBf, 16384 / 8);
  cvt_bf16_kernel<<<blocks(16384 / 8, T), T, 0, stream>>>(W2r, W2rBf, 16384 / 8);
  cvt_bf16_kernel<<<blocks(16384 / 8, T), T, 0, stream>>>(W3l, W3lBf, 16384 / 8);
  cvt_bf16_kernel<<<blocks(16384 / 8, T), T, 0, stream>>>(W3r, W3rBf, 16384 / 8);
  cvt_bf16_kernel<<<blocks(65536 / 8, T), T, 0, stream>>>(Wlin, WlinBf, 65536 / 8);
  cvt_bf16_kernel<<<blocks(featN / 8, T), T, 0, stream>>>(x, xBf, featN / 8);

  // degrees
  fill_u32_kernel<<<blocks((size_t)N, T), T, 0, stream>>>((uint32_t*)deg, 0u, (size_t)N);
  degree_kernel<<<blocks((size_t)E, T), T, 0, stream>>>(dst, deg, E);

  const unsigned scatterB = blocks((size_t)E * 32, T);
  const unsigned gemmB = blocks((size_t)((N + 15) / 16) * 8, 8);  // waves/8 per block

  // layer 1: x -> F0 (relu), bf16 copy in h1Bf
  fill_u32_kernel<<<blocks(featN, T), T, 0, stream>>>((uint32_t*)agg, 0u, featN);
  scatter_add_kernel<<<scatterB, T, 0, stream>>>(x, src, dst, agg, E);
  cvt_scale_bf16_kernel<<<blocks(featN / 8, T), T, 0, stream>>>(agg, deg, aggBf, featN / 8);
  sage_gemm_kernel<<<gemmB, T, 0, stream>>>(aggBf, xBf, W1lBf, W1rBf, b1, F0, h1Bf, N, 1);

  // layer 2: F0 -> F0 (relu), bf16 copy in h2Bf (aliases xBf, now dead)
  fill_u32_kernel<<<blocks(featN, T), T, 0, stream>>>((uint32_t*)agg, 0u, featN);
  scatter_add_kernel<<<scatterB, T, 0, stream>>>(F0, src, dst, agg, E);
  cvt_scale_bf16_kernel<<<blocks(featN / 8, T), T, 0, stream>>>(agg, deg, aggBf, featN / 8);
  sage_gemm_kernel<<<gemmB, T, 0, stream>>>(aggBf, h1Bf, W2lBf, W2rBf, b2, F0, h2Bf, N, 1);

  // layer 3: F0 -> F0 (no relu, no bf16 output needed)
  fill_u32_kernel<<<blocks(featN, T), T, 0, stream>>>((uint32_t*)agg, 0u, featN);
  scatter_add_kernel<<<scatterB, T, 0, stream>>>(F0, src, dst, agg, E);
  cvt_scale_bf16_kernel<<<blocks(featN / 8, T), T, 0, stream>>>(agg, deg, aggBf, featN / 8);
  sage_gemm_kernel<<<gemmB, T, 0, stream>>>(aggBf, h2Bf, W3lBf, W3rBf, b3, F0,
                                            ((__bf16*)nullptr), N, 0);

  // pooling
  fill_u32_kernel<<<blocks((size_t)G * 128, T), T, 0, stream>>>((uint32_t*)maxk, NEG_INF_KEY, (size_t)G * 128);
  fill_u32_kernel<<<blocks((size_t)G * 128, T), T, 0, stream>>>((uint32_t*)sums, 0u, (size_t)G * 128);
  fill_u32_kernel<<<blocks((size_t)G, T), T, 0, stream>>>((uint32_t*)cnt, 0u, (size_t)G);
  pool_scatter_kernel<<<blocks(featN, T), T, 0, stream>>>(F0, batch, maxk, sums, cnt, N);
  build_pooled_kernel<<<blocks((size_t)G * 256, T), T, 0, stream>>>(maxk, sums, cnt, pooledBf, G);

  // final linear head -> d_out
  const unsigned finWaves = (unsigned)(((G + 15) / 16) * 16);
  final_gemm_kernel<<<(finWaves + 7) / 8, T, 0, stream>>>(pooledBf, WlinBf, blin, outp, G);
}
